// ContrastiveLoss_25366076850213
// MI455X (gfx1250) — compile-verified, matrix operand order checked by reading, not run on hardware
//
#include <hip/hip_runtime.h>
#include <hip/hip_bf16.h>
#include <cstdint>
#include <cstddef>

typedef _Float16 half8   __attribute__((ext_vector_type(8)));
typedef _Float16 half4_t __attribute__((ext_vector_type(4)));
typedef _Float16 v16h    __attribute__((ext_vector_type(16)));
typedef float    v8f     __attribute__((ext_vector_type(8)));

#define CN 8192          // rows per view
#define CD 1024          // feature dim
#define SPLITS 32        // column splits per M-block (one wave each)
#define NBT 4            // 64-wide column blocks per wave (32*4*64 = 8192)
#define TEMP_INV 2.0f    // 1 / 0.5
#define MAXB 2.0f        // fixed logsumexp shift: |cos/T| <= 2 for unit rows

// ---------------- Kernel 1: L2-normalize rows, fp32 -> f16 ----------------
__global__ __launch_bounds__(256) void nrm_f16_kernel(const float* __restrict__ v1,
                                                      const float* __restrict__ v2,
                                                      _Float16* __restrict__ z1,
                                                      _Float16* __restrict__ z2) {
  int row = blockIdx.x;                       // 0 .. 2*CN-1
  const float* src;
  _Float16* dst;
  if (row < CN) { src = v1 + (size_t)row * CD;        dst = z1 + (size_t)row * CD; }
  else          { int r = row - CN;
                  src = v2 + (size_t)r * CD;          dst = z2 + (size_t)r * CD; }
  int t = threadIdx.x;                        // 256 threads, 4 floats each
  float4 x = ((const float4*)src)[t];
  float ss = x.x * x.x + x.y * x.y + x.z * x.z + x.w * x.w;
  #pragma unroll
  for (int m = 1; m < 32; m <<= 1) ss += __shfl_xor(ss, m, 32);
  __shared__ float red[8];
  if ((t & 31) == 0) red[t >> 5] = ss;
  __syncthreads();
  float tot = 0.f;
  #pragma unroll
  for (int i = 0; i < 8; ++i) tot += red[i];
  float scale = 1.0f / fmaxf(sqrtf(tot), 1e-12f);
  half4_t h;
  h[0] = (_Float16)(x.x * scale);
  h[1] = (_Float16)(x.y * scale);
  h[2] = (_Float16)(x.z * scale);
  h[3] = (_Float16)(x.w * scale);
  ((half4_t*)dst)[t] = h;
}

// ------------- Kernel 2: diagonal sims, one wave per row -------------
__global__ __launch_bounds__(256) void diag_kernel(const _Float16* __restrict__ z1,
                                                   const _Float16* __restrict__ z2,
                                                   float* __restrict__ diag) {
  int wave = threadIdx.x >> 5;
  int lane = threadIdx.x & 31;
  int row  = blockIdx.x * 8 + wave;
  const _Float16* p = z1 + (size_t)row * CD;
  const _Float16* q = z2 + (size_t)row * CD;
  float acc = 0.f;
  for (int c = lane; c < CD / 8; c += 32) {    // 4 x half8 per lane
    half8 a = *(const half8*)(p + c * 8);
    half8 b = *(const half8*)(q + c * 8);
    #pragma unroll
    for (int i = 0; i < 8; ++i) acc += (float)a[i] * (float)b[i];
  }
  #pragma unroll
  for (int m = 1; m < 32; m <<= 1) acc += __shfl_xor(acc, m, 32);
  if (lane == 0) diag[row] = acc * TEMP_INV;
}

// -------- Kernel 3: fused WMMA GEMM (z1 · z2^T / T) + fixed-shift sumexp --------
// grid = (CN/32 M-blocks, SPLITS/4), block = 128 (4 waves).
// Wave w handles column split = blockIdx.y*4 + w: NBT blocks of 64 columns.
// Each wave register-blocks a 32x64 output (2x4 WMMA tiles).
__global__ __launch_bounds__(128) void gemm_sumexp_kernel(const _Float16* __restrict__ z1,
                                                          const _Float16* __restrict__ z2,
                                                          float* __restrict__ partials) {
  __shared__ _Float16 Atile[32 * CD] __attribute__((aligned(16)));  // 64 KB
  int mblk = blockIdx.x;                       // 32-row block index
  const _Float16* Ag = z1 + (size_t)mblk * 32 * CD;
  for (int c = threadIdx.x; c < 32 * CD / 8; c += 128)
    *(half8*)(Atile + c * 8) = *(const half8*)(Ag + c * 8);
  __syncthreads();

  int wave = threadIdx.x >> 5;
  int lane = threadIdx.x & 31;
  int hf   = lane >> 4;        // lane half: selects K sub-chunk
  int mn   = lane & 15;        // A: row-in-tile / B: col-in-tile / C: column n
  int split = blockIdx.y * 4 + wave;           // 0..31

  float ssum[2][8];
  #pragma unroll
  for (int im = 0; im < 2; ++im)
    #pragma unroll
    for (int r = 0; r < 8; ++r) ssum[im][r] = 0.f;

  // A fragments: Mtile im -> LDS row im*16 + mn
  const _Float16* Arow = Atile + mn * CD + hf * 8;

  for (int t = 0; t < NBT; ++t) {
    int nb = split * NBT + t;                  // 64-wide column block, 0..127
    const _Float16* Brow = z2 + ((size_t)nb * 64 + mn) * CD + hf * 8;
    v8f acc[2][4];
    #pragma unroll
    for (int im = 0; im < 2; ++im)
      #pragma unroll
      for (int in = 0; in < 4; ++in) acc[im][in] = (v8f){};

    for (int kc = 0; kc < CD / 32; ++kc) {
      int kb = kc * 32;
      v16h av[2], bv[4];
      #pragma unroll
      for (int im = 0; im < 2; ++im) {
        half8 lo = *(const half8*)(Arow + im * 16 * CD + kb);        // ds_load_b128
        half8 hi = *(const half8*)(Arow + im * 16 * CD + kb + 16);
        #pragma unroll
        for (int i = 0; i < 8; ++i) { av[im][i] = lo[i]; av[im][i + 8] = hi[i]; }
      }
      #pragma unroll
      for (int in = 0; in < 4; ++in) {
        half8 lo = *(const half8*)(Brow + (size_t)in * 16 * CD + kb); // global_load_b128
        half8 hi = *(const half8*)(Brow + (size_t)in * 16 * CD + kb + 16);
        #pragma unroll
        for (int i = 0; i < 8; ++i) { bv[in][i] = lo[i]; bv[in][i + 8] = hi[i]; }
      }
      #pragma unroll
      for (int im = 0; im < 2; ++im)
        #pragma unroll
        for (int in = 0; in < 4; ++in)
          acc[im][in] = __builtin_amdgcn_wmma_f32_16x16x32_f16(
              false, av[im], false, bv[in], (short)0, acc[im][in], false, false);
    }
    // sumexp with fixed shift: v = acc*2 in [-2,2]  =>  exp(v - 2) safe
    #pragma unroll
    for (int im = 0; im < 2; ++im)
      #pragma unroll
      for (int in = 0; in < 4; ++in)
        #pragma unroll
        for (int r = 0; r < 8; ++r)
          ssum[im][r] += __expf(acc[im][in][r] * TEMP_INV - MAXB);
  }

  // sum across the 16 lanes of each half (same row, 16 columns each)
  #pragma unroll
  for (int im = 0; im < 2; ++im)
    #pragma unroll
    for (int r = 0; r < 8; ++r) {
      float s = ssum[im][r];
      #pragma unroll
      for (int msk = 1; msk <= 8; msk <<= 1) s += __shfl_xor(s, msk, 32);
      ssum[im][r] = s;
    }

  if (mn == 0) {                 // lanes 0 and 16: rows 0..7 / 8..15 of each tile
    #pragma unroll
    for (int im = 0; im < 2; ++im) {
      int rowbase = mblk * 32 + im * 16 + 8 * hf;
      #pragma unroll
      for (int r = 0; r < 8; ++r)
        partials[(size_t)(rowbase + r) * SPLITS + split] = ssum[im][r];
    }
  }
}

// ---------------- Kernel 4: combine splits -> per-row loss ----------------
__global__ __launch_bounds__(256) void lse_finalize_kernel(const float* __restrict__ partials,
                                                           const float* __restrict__ diag,
                                                           float* __restrict__ rowloss) {
  int row = blockIdx.x * 256 + threadIdx.x;
  float s = 0.f;
  for (int k = 0; k < SPLITS; ++k) s += partials[(size_t)row * SPLITS + k];
  rowloss[row] = MAXB + __logf(s) - diag[row];   // lse_i - sim_ii
}

// ------------- Kernel 5: deterministic mean over rows -> out[0] -------------
__global__ __launch_bounds__(256) void mean_reduce_kernel(const float* __restrict__ rowloss,
                                                          float* __restrict__ out) {
  __shared__ float red[256];
  int t = threadIdx.x;
  float s = 0.f;
  for (int i = t; i < CN; i += 256) s += rowloss[i];   // fixed order
  red[t] = s;
  __syncthreads();
  for (int off = 128; off > 0; off >>= 1) {
    if (t < off) red[t] += red[t + off];
    __syncthreads();
  }
  if (t == 0) out[0] = red[0] / (float)CN;
}

extern "C" void kernel_launch(void* const* d_in, const int* in_sizes, int n_in,
                              void* d_out, int out_size, void* d_ws, size_t ws_size,
                              hipStream_t stream) {
  const float* v1 = (const float*)d_in[0];
  const float* v2 = (const float*)d_in[1];
  float* out = (float*)d_out;

  char* ws = (char*)d_ws;
  _Float16* z1      = (_Float16*)(ws);                     // 16 MB
  _Float16* z2      = (_Float16*)(ws + 16777216);          // 16 MB
  float*    partial = (float*)   (ws + 33554432);          // 8192*32 f32 = 1 MB
  float*    diag    = (float*)   (ws + 34603008);          // 32 KB
  float*    rowloss = (float*)   (ws + 34635776);          // 32 KB

  nrm_f16_kernel<<<2 * CN, 256, 0, stream>>>(v1, v2, z1, z2);
  diag_kernel<<<CN / 8, 256, 0, stream>>>(z1, z2, diag);
  gemm_sumexp_kernel<<<dim3(CN / 32, SPLITS / 4), 128, 0, stream>>>(z1, z2, partial);
  lse_finalize_kernel<<<CN / 256, 256, 0, stream>>>(partial, diag, rowloss);
  mean_reduce_kernel<<<1, 256, 0, stream>>>(rowloss, out);
}
